// BertSelfAttention2_59236188946617
// MI455X (gfx1250) — matrix operand order repaired and checked by reading, not compiled, not run on hardware
//
#include <hip/hip_runtime.h>
#include <hip/hip_bf16.h>
#include <math.h>

#define NTOK   4096   // B*S
#define DMODEL 1024
#define SEQ    2048
#define NHEAD  16
#define HD     64

typedef float  v8f   __attribute__((ext_vector_type(8)));
typedef __bf16 v16bf __attribute__((ext_vector_type(16)));
typedef __bf16 v8bf  __attribute__((ext_vector_type(8)));
typedef int    v4i   __attribute__((vector_size(16)));

#define __AS(n) __attribute__((address_space(n)))

#if __has_builtin(__builtin_amdgcn_global_load_async_to_lds_b128) && \
    __has_builtin(__builtin_amdgcn_s_wait_asynccnt)
#define HAVE_ASYNC 1
#else
#define HAVE_ASYNC 0
#endif

// 16-byte global -> LDS copy. Async (ASYNCcnt) when available.
static __device__ __forceinline__ void cp16(void* lds, const void* g) {
#if HAVE_ASYNC
  __builtin_amdgcn_global_load_async_to_lds_b128(
      (__AS(1) v4i*)g, (__AS(3) v4i*)lds, 0, 0);
#else
  *(v8bf*)lds = *(const v8bf*)g;
#endif
}

template <int N>
static __device__ __forceinline__ void async_wait() {
#if HAVE_ASYNC
  __builtin_amdgcn_s_wait_asynccnt(N);
#endif
}

static __device__ __forceinline__ v16bf cat16(v8bf lo, v8bf hi) {
  return __builtin_shufflevector(lo, hi, 0,1,2,3,4,5,6,7,8,9,10,11,12,13,14,15);
}

static __device__ __forceinline__ v8f wmma_bf16(v16bf a, v16bf b, v8f c) {
  // D = A(16x32 bf16) * B(32x16 bf16) + C(16x16 f32)
  return __builtin_amdgcn_wmma_f32_16x16x32_bf16(false, a, false, b, (short)0, c,
                                                 false, false);
}

static __device__ __forceinline__ float halfmax(float x) {
#pragma unroll
  for (int o = 1; o < 16; o <<= 1) x = fmaxf(x, __shfl_xor(x, o, 32));
  return x;
}
static __device__ __forceinline__ float halfsum(float x) {
#pragma unroll
  for (int o = 1; o < 16; o <<= 1) x += __shfl_xor(x, o, 32);
  return x;
}

// ---------------------------------------------------------------------------
// Prep A: X f32 -> bf16 (one time). 8 elements per thread.
// ---------------------------------------------------------------------------
__global__ __launch_bounds__(256) void cvt_x_kernel(
    const float* __restrict__ X, __bf16* __restrict__ Xb)
{
  size_t i = ((size_t)blockIdx.x * 256 + threadIdx.x) * 8;
  float4 a = *(const float4*)(X + i);
  float4 b = *(const float4*)(X + i + 4);
  v8bf o = {(__bf16)a.x, (__bf16)a.y, (__bf16)a.z, (__bf16)a.w,
            (__bf16)b.x, (__bf16)b.y, (__bf16)b.z, (__bf16)b.w};
  *(v8bf*)(Xb + i) = o;
}

// ---------------------------------------------------------------------------
// Prep B: W [K][N] f32 -> Wt [N][K] bf16 (one time). 64x64 tile via LDS.
// blockIdx.z selects Wq (0) / Wk (1).
// ---------------------------------------------------------------------------
__global__ __launch_bounds__(256) void transpose_w_kernel(
    const float* __restrict__ Wq, const float* __restrict__ Wk,
    __bf16* __restrict__ Wqt, __bf16* __restrict__ Wkt)
{
  __shared__ __align__(16) __bf16 T[64][64];
  const float* W  = blockIdx.z ? Wk  : Wq;
  __bf16*      Wt = blockIdx.z ? Wkt : Wqt;
  const int kB = blockIdx.x * 64;
  const int nB = blockIdx.y * 64;
  const int t = threadIdx.x;
#pragma unroll
  for (int i = 0; i < 4; ++i) {            // 64x64 f32 = 1024 float4
    int id = t + i * 256;
    int kr = id >> 4, c4 = (id & 15) << 2;
    float4 f = *(const float4*)(W + (size_t)(kB + kr) * DMODEL + nB + c4);
    T[c4 + 0][kr] = (__bf16)f.x; T[c4 + 1][kr] = (__bf16)f.y;
    T[c4 + 2][kr] = (__bf16)f.z; T[c4 + 3][kr] = (__bf16)f.w;
  }
  __syncthreads();
#pragma unroll
  for (int i = 0; i < 2; ++i) {            // 64 rows x 8 chunks of 16B
    int id = t + i * 256;
    int n = id >> 3, c = (id & 7) << 3;
    *(v8bf*)(Wt + (size_t)(nB + n) * DMODEL + kB + c) = *(const v8bf*)&T[n][c];
  }
}

// ---------------------------------------------------------------------------
// Kernel 1: fused Q/K projection GEMM + bias + (v = q+k, kt = softplus(k)).
// bf16 operands pre-staged; LDS filled by async b128 copies, double-buffered.
// Tile 64x64, K-step 32; 4 waves, each 32x32 (2x2 WMMA) for q AND k.
// v written pre-transposed: vT[b][d][s].
// ---------------------------------------------------------------------------
__global__ __launch_bounds__(128) void qk_proj_kernel(
    const __bf16* __restrict__ Xb,
    const __bf16* __restrict__ Wqt, const float* __restrict__ bq,
    const __bf16* __restrict__ Wkt, const float* __restrict__ bk,
    __bf16* __restrict__ q_out, __bf16* __restrict__ kt_out,
    __bf16* __restrict__ vT)
{
  __shared__ __align__(16) __bf16 Xs [2][64][32];
  __shared__ __align__(16) __bf16 Wqs[2][64][32];
  __shared__ __align__(16) __bf16 Wks[2][64][32];

  const int mBase = blockIdx.x * 64;
  const int nBase = blockIdx.y * 64;
  const int t = threadIdx.x, lane = t & 31, wave = t >> 5;
  const int wm = (wave >> 1) * 32, wn = (wave & 1) * 32;
  const int l16 = lane & 15, hl = lane >> 4;

  v8f accq[2][2] = {};
  v8f acck[2][2] = {};

  auto issue = [&](int kb, int bb) {
#pragma unroll
    for (int i = 0; i < 2; ++i) {          // 64 rows x 4 chunks, 2 per thread
      int id = t + i * 128;
      int r = id >> 2, c = (id & 3) << 3;  // chunk = 8 bf16 = 16 B
      cp16(&Xs [bb][r][c], Xb  + (size_t)(mBase + r) * DMODEL + kb + c);
      cp16(&Wqs[bb][r][c], Wqt + (size_t)(nBase + r) * DMODEL + kb + c);
      cp16(&Wks[bb][r][c], Wkt + (size_t)(nBase + r) * DMODEL + kb + c);
    }
  };

  issue(0, 0);
  for (int it = 0; it < DMODEL / 32; ++it) {
    if (it + 1 < DMODEL / 32) {
      issue((it + 1) * 32, (it + 1) & 1);  // prefetch next tile (other buffer)
      async_wait<6>();                     // this tile's 6 copies done (in-order)
    } else {
      async_wait<0>();
    }
    __syncthreads();
    const int cur = it & 1;

    // A fragments: lane row M = l16, K chunks at hl*8 and 16+hl*8.
    v16bf a[2], bqf[2], bkf[2];
#pragma unroll
    for (int mi = 0; mi < 2; ++mi) {
      int r = wm + mi * 16 + l16;
      a[mi] = cat16(*(const v8bf*)&Xs[cur][r][hl * 8],
                    *(const v8bf*)&Xs[cur][r][16 + hl * 8]);
    }
    // B fragments: lane col N = l16, K chunks at hl*8 and 16+hl*8.
#pragma unroll
    for (int ni = 0; ni < 2; ++ni) {
      int n = wn + ni * 16 + l16;
      bqf[ni] = cat16(*(const v8bf*)&Wqs[cur][n][hl * 8],
                      *(const v8bf*)&Wqs[cur][n][16 + hl * 8]);
      bkf[ni] = cat16(*(const v8bf*)&Wks[cur][n][hl * 8],
                      *(const v8bf*)&Wks[cur][n][16 + hl * 8]);
    }
#pragma unroll
    for (int mi = 0; mi < 2; ++mi)
#pragma unroll
      for (int ni = 0; ni < 2; ++ni) {
        accq[mi][ni] = wmma_bf16(a[mi], bqf[ni], accq[mi][ni]);
        acck[mi][ni] = wmma_bf16(a[mi], bkf[ni], acck[mi][ni]);
      }
    __syncthreads();                       // reads done before buffer reuse
  }

  // Epilogue: bias, v = q+k, kt = softplus(k).
#pragma unroll
  for (int ni = 0; ni < 2; ++ni) {
    int n = nBase + wn + ni * 16 + l16;
    float biasq = bq[n];
    float biask = bk[n];
#pragma unroll
    for (int mi = 0; mi < 2; ++mi) {
#pragma unroll
      for (int j = 0; j < 8; ++j) {
        int m = mBase + wm + mi * 16 + j + 8 * hl;
        float qv = accq[mi][ni][j] + biasq;
        float kv = acck[mi][ni][j] + biask;
        float vv = qv + kv;
        float sp = fmaxf(kv, 0.0f) + log1pf(__expf(-fabsf(kv)));  // softplus
        size_t idx = (size_t)m * DMODEL + n;
        q_out[idx]  = (__bf16)qv;
        kt_out[idx] = (__bf16)sp;
        int bi = m >> 11, s = m & (SEQ - 1);
        vT[((size_t)bi * DMODEL + n) * SEQ + s] = (__bf16)vv;  // transposed
      }
    }
  }
}

// ---------------------------------------------------------------------------
// Kernel 2: flash attention per (b, h). Block = 4 waves x 16 q-rows.
// kt tile row-major [key][d] (B layout for q.kt^T); v tile from pre-transposed
// vT (B layout for probs.v). Async double-buffered LDS tiles.
// ---------------------------------------------------------------------------
__global__ __launch_bounds__(128) void attn_kernel(
    const __bf16* __restrict__ q_bf, const __bf16* __restrict__ kt_bf,
    const __bf16* __restrict__ vT, const int* __restrict__ amask,
    float* __restrict__ out)
{
  __shared__ __align__(16) __bf16 Ks[2][32][64];
  __shared__ __align__(16) __bf16 Vt[2][64][32];
  __shared__ __align__(16) __bf16 Ps[4][16][32];

  const int b    = blockIdx.z;
  const int h    = blockIdx.y;
  const int qT   = blockIdx.x;
  const int t    = threadIdx.x;
  const int lane = t & 31, wave = t >> 5;
  const int l16  = lane & 15, hl = lane >> 4;
  const int tokBase = b * SEQ;
  const int qRow0   = qT * 64 + wave * 16;
  const int hcol    = h * HD;

  auto issue = [&](int kb, int bb) {
#pragma unroll
    for (int i = 0; i < 2; ++i) {          // Ks: 32 keys x 8 chunks
      int id = t + i * 128;
      int r = id >> 3, c = (id & 7) << 3;
      cp16(&Ks[bb][r][c],
           kt_bf + (size_t)(tokBase + kb + r) * DMODEL + hcol + c);
    }
#pragma unroll
    for (int i = 0; i < 2; ++i) {          // Vt: 64 d-rows x 4 chunks
      int id = t + i * 128;
      int r = id >> 2, c = (id & 3) << 3;
      cp16(&Vt[bb][r][c],
           vT + ((size_t)b * DMODEL + hcol + r) * SEQ + kb + c);
    }
  };

  // Q fragments for this wave's 16 rows: K(d) chunks 0..31 and 32..63.
  v16bf qa[2];
  {
    const __bf16* qp = q_bf + (size_t)(tokBase + qRow0 + l16) * DMODEL + hcol;
#pragma unroll
    for (int f = 0; f < 2; ++f) {
      qa[f] = cat16(*(const v8bf*)(qp + f * 32 + hl * 8),
                    *(const v8bf*)(qp + f * 32 + 16 + hl * 8));
    }
  }

  float mrow[8], lrow[8];
  int   mz[8];
#pragma unroll
  for (int j = 0; j < 8; ++j) {
    int m = qRow0 + j + 8 * hl;
    mz[j]   = (amask[tokBase + m] == 0);   // query-axis mask
    mrow[j] = -3.0e38f;
    lrow[j] = 0.0f;
  }
  v8f acc[4] = {};

  issue(0, 0);
  for (int it = 0; it < SEQ / 32; ++it) {
    if (it + 1 < SEQ / 32) {
      issue((it + 1) * 32, (it + 1) & 1);
      async_wait<4>();
    } else {
      async_wait<0>();
    }
    __syncthreads();
    const int cur = it & 1;

    // scores: two 16-key subtiles, each = 2 WMMAs over d (K=32 each)
    v8f sc[2];
#pragma unroll
    for (int kk = 0; kk < 2; ++kk) {
      int n = kk * 16 + l16;
      v8f s = {};
#pragma unroll
      for (int f = 0; f < 2; ++f) {
        s = wmma_bf16(qa[f],
                      cat16(*(const v8bf*)&Ks[cur][n][f * 32 + hl * 8],
                            *(const v8bf*)&Ks[cur][n][f * 32 + 16 + hl * 8]),
                      s);
      }
      sc[kk] = s;
    }

    // online softmax over the 32-key tile (rows M = j + 8*hl, col = lane%16)
    float alpha[8];
#pragma unroll
    for (int j = 0; j < 8; ++j) {
      float x0 = mz[j] ? -10000.0f : sc[0][j] * 0.125f;  // 1/sqrt(64)
      float x1 = mz[j] ? -10000.0f : sc[1][j] * 0.125f;
      float tm   = halfmax(fmaxf(x0, x1));
      float mnew = fmaxf(mrow[j], tm);
      float p0 = __expf(x0 - mnew);
      float p1 = __expf(x1 - mnew);
      float rs = halfsum(p0 + p1);
      float al = __expf(mrow[j] - mnew);
      lrow[j] = lrow[j] * al + rs;
      mrow[j] = mnew;
      alpha[j] = al;
      int M = j + 8 * hl;
      Ps[wave][M][l16]      = (__bf16)p0;  // C-layout -> A-layout via LDS
      Ps[wave][M][16 + l16] = (__bf16)p1;
    }
#pragma unroll
    for (int g = 0; g < 4; ++g)
#pragma unroll
      for (int j = 0; j < 8; ++j) acc[g][j] *= alpha[j];

    // probs as A fragment (16 q-rows x 32 keys); DS in-order within wave.
    v16bf pa = cat16(*(const v8bf*)&Ps[wave][l16][hl * 8],
                     *(const v8bf*)&Ps[wave][l16][16 + hl * 8]);
    // context: 4 WMMAs over d-groups, K = 32 keys
#pragma unroll
    for (int g = 0; g < 4; ++g) {
      acc[g] = wmma_bf16(pa,
                         cat16(*(const v8bf*)&Vt[cur][g * 16 + l16][hl * 8],
                               *(const v8bf*)&Vt[cur][g * 16 + l16][16 + hl * 8]),
                         acc[g]);
    }
    __syncthreads();
  }

  // normalize and write [B,S,D] f32
#pragma unroll
  for (int g = 0; g < 4; ++g) {
#pragma unroll
    for (int j = 0; j < 8; ++j) {
      int m = qRow0 + j + 8 * hl;
      out[(size_t)(tokBase + m) * DMODEL + hcol + g * 16 + l16] =
          acc[g][j] / lrow[j];
    }
  }
}

// ---------------------------------------------------------------------------
extern "C" void kernel_launch(void* const* d_in, const int* in_sizes, int n_in,
                              void* d_out, int out_size, void* d_ws, size_t ws_size,
                              hipStream_t stream) {
  const float* X     = (const float*)d_in[0];
  const int*   amask = (const int*)d_in[1];
  const float* Wq    = (const float*)d_in[2];
  const float* bq    = (const float*)d_in[3];
  const float* Wk    = (const float*)d_in[4];
  const float* bk    = (const float*)d_in[5];
  float* out = (float*)d_out;

  __bf16* Xb    = (__bf16*)d_ws;                          // 8 MB
  __bf16* Wqt   = Xb  + (size_t)NTOK * DMODEL;            // 2 MB
  __bf16* Wkt   = Wqt + (size_t)DMODEL * DMODEL;          // 2 MB
  __bf16* q_bf  = Wkt + (size_t)DMODEL * DMODEL;          // 8 MB
  __bf16* kt_bf = q_bf + (size_t)NTOK * DMODEL;           // 8 MB
  __bf16* vT    = kt_bf + (size_t)NTOK * DMODEL;          // 8 MB

  cvt_x_kernel<<<(NTOK * DMODEL) / (256 * 8), 256, 0, stream>>>(X, Xb);
  dim3 gT(DMODEL / 64, DMODEL / 64, 2);
  transpose_w_kernel<<<gT, 256, 0, stream>>>(Wq, Wk, Wqt, Wkt);

  dim3 g1(NTOK / 64, DMODEL / 64, 1);      // 64 x 16 blocks
  qk_proj_kernel<<<g1, 128, 0, stream>>>(Xb, Wqt, bq, Wkt, bk, q_bf, kt_bf, vT);

  dim3 g2(SEQ / 64, NHEAD, 2);             // 32 x 16 x 2 blocks
  attn_kernel<<<g2, 128, 0, stream>>>(q_bf, kt_bf, vT, amask, out);
}
